// DomainAwareSchNet_27041114095755
// MI455X (gfx1250) — matrix operand order repaired and checked by reading, not compiled, not run on hardware
//
#include <hip/hip_runtime.h>
#include <math.h>

// ---------------------------------------------------------------------------
// DomainAwareSchNet for MI455X (gfx1250, wave32, WMMA bf16)
// ---------------------------------------------------------------------------

typedef __bf16 bf16;
typedef __attribute__((ext_vector_type(16))) __bf16 v16bf;
typedef __attribute__((ext_vector_type(8)))  float  v8f;

#define NATOMS  8192
#define NEDGES  262144
#define NGRAPH  32
#define KNEI    32
#define HD      128
#define NGPAD   64      // NG=50 padded to 64 (zeros) for K-multiple-of-32
#define NLAYERS 6

#if defined(__HIP_DEVICE_COMPILE__) && \
    __has_builtin(__builtin_amdgcn_global_load_async_to_lds_b64) && \
    __has_builtin(__builtin_amdgcn_s_wait_asynccnt)
#define HAVE_ASYNC_LDS 1
typedef int v2i __attribute__((ext_vector_type(2)));
typedef __attribute__((address_space(1))) v2i gv2i;   // global AS pointer pointee
typedef __attribute__((address_space(3))) v2i lv2i;   // LDS AS pointer pointee
#else
#define HAVE_ASYNC_LDS 0
#endif

__device__ __forceinline__ float sspf(float x) {
  // shifted softplus: log(1+exp(x)) - log(2), numerically stable
  float sp = (x > 20.f) ? x : log1pf(__expf(x));
  return sp - 0.6931471805599453f;
}

__device__ __forceinline__ v8f wmma_bf(v16bf a, v16bf b, v8f c) {
  // D = A(16x32 bf16) * B(32x16 bf16) + C(16x16 f32)
  return __builtin_amdgcn_wmma_f32_16x16x32_bf16(
      /*neg_a=*/false, a, /*neg_b=*/false, b,
      /*c_mod=*/(short)0, c, /*reuse_a=*/false, /*reuse_b=*/false);
}

// A fragment (16x32, bf16, row-major source, rows m0..m0+15, cols k0..k0+31)
// ISA layout: lane m = lane%16, half = lane/16; vgpr v holds K pair
//   k = (v<4 ? 0 : 16) + half*8 + (v&3)*2
__device__ __forceinline__ v16bf load_a_bf16(const bf16* A, int ld, int m0,
                                             int k0, int lane) {
  int m = m0 + (lane & 15);
  int half = lane >> 4;
  const bf16* row = A + (long)m * ld + k0;
  union { v16bf v; unsigned u[8]; } r;
#pragma unroll
  for (int v = 0; v < 8; ++v) {
    int kk = ((v < 4) ? 0 : 16) + half * 8 + (v & 3) * 2;
    r.u[v] = *(const unsigned*)(row + kk);
  }
  return r.v;
}

// A fragment loaded from f32 source with in-register f32->bf16 conversion
__device__ __forceinline__ v16bf load_a_f32(const float* A, int ld, int m0,
                                            int k0, int lane) {
  int m = m0 + (lane & 15);
  int half = lane >> 4;
  const float* row = A + (long)m * ld + k0;
  union { v16bf v; bf16 h[16]; } r;
#pragma unroll
  for (int v = 0; v < 8; ++v) {
    int kk = ((v < 4) ? 0 : 16) + half * 8 + (v & 3) * 2;
    r.h[2 * v]     = (bf16)row[kk];
    r.h[2 * v + 1] = (bf16)row[kk + 1];
  }
  return r.v;
}

// B fragment (32x16 bf16, row-major KxN source): lane%16 = N col, 16 halves = K (same order as A)
__device__ __forceinline__ v16bf load_b_bf16(const bf16* B, int ld, int k0,
                                             int n0, int lane) {
  int half = lane >> 4;
  const bf16* col = B + (long)k0 * ld + n0 + (lane & 15);
  union { v16bf v; bf16 h[16]; } r;
#pragma unroll
  for (int s = 0; s < 16; ++s) {
    int kk = ((s < 8) ? 0 : 16) + half * 8 + (s & 7);
    r.h[s] = col[(long)kk * ld];
  }
  return r.v;
}

// prefetch a weight matrix (nbytes) spread across the block's threads
__device__ __forceinline__ void prefetch_weights(const void* p, int nbytes) {
  int lines = nbytes >> 7;                     // 128B cachelines
  for (int t = threadIdx.x; t < lines; t += blockDim.x)
    __builtin_prefetch((const char*)p + (t << 7), 0, 3);
}

// ---------------------------------------------------------------------------
// small elementwise kernels
// ---------------------------------------------------------------------------

__global__ void k_zero(float* out, int n) {
  int i = blockIdx.x * blockDim.x + threadIdx.x;
  if (i < n) out[i] = 0.f;
}

__global__ void k_convert_bf16(const float* in, bf16* out, int n) {
  int i = blockIdx.x * blockDim.x + threadIdx.x;
  if (i < n) out[i] = (bf16)in[i];
}

// mlp_w1 [L,50,128] -> padded [L,64,128] bf16 (zero rows 50..63)
__global__ void k_pad_w1(const float* in, bf16* out) {
  int i = blockIdx.x * blockDim.x + threadIdx.x;   // over L*64*128
  int l = i / (NGPAD * HD);
  int rem = i - l * NGPAD * HD;
  int k = rem / HD, n = rem - k * HD;
  out[i] = (k < 50) ? (bf16)in[(l * 50 + k) * HD + n] : (bf16)0.f;
}

// h = embedding[z]
__global__ void k_embed(const int* z, const float* emb, float* h) {
  int i = blockIdx.x * blockDim.x + threadIdx.x;   // over NATOMS*HD
  int n = i >> 7, c = i & 127;
  h[i] = emb[z[n] * HD + c];
}

// per-edge: distance -> gaussian basis (bf16, padded to 64) + cosine cutoff
__global__ void k_prep_edges(const float* pos, const int* ei, bf16* ea,
                             float* Cbuf) {
  int e = blockIdx.x * blockDim.x + threadIdx.x;
  if (e >= NEDGES) return;
  int s = ei[e], t = ei[NEDGES + e];
  float dx = pos[3 * s] - pos[3 * t];
  float dy = pos[3 * s + 1] - pos[3 * t + 1];
  float dz = pos[3 * s + 2] - pos[3 * t + 2];
  float d = sqrtf(dx * dx + dy * dy + dz * dz);
  Cbuf[e] = 0.5f * (__cosf(d * 0.31415926535f) + 1.f);
  const float step = 10.f / 49.f;
  const float coeff = -0.5f / (step * step);
  bf16* row = ea + (long)e * NGPAD;
#pragma unroll 8
  for (int g = 0; g < NGPAD; g += 2) {   // packed dword stores
    float v0 = (g < 50) ? __expf(coeff * (d - step * g) * (d - step * g)) : 0.f;
    float v1 = (g + 1 < 50)
                   ? __expf(coeff * (d - step * (g + 1)) * (d - step * (g + 1)))
                   : 0.f;
    union { unsigned u; bf16 h[2]; } pk;
    pk.h[0] = (bf16)v0;
    pk.h[1] = (bf16)v1;
    *(unsigned*)(row + g) = pk.u;
  }
}

// ---------------------------------------------------------------------------
// node GEMM: xg = bf16( h[8192,128] @ W[128,128] )   (wave = 16-row strip)
// ---------------------------------------------------------------------------
__global__ __launch_bounds__(256) void k_node_gemm(const float* h,
                                                   const bf16* W, bf16* xg) {
  prefetch_weights(W, HD * HD * 2);
  int wave = threadIdx.x >> 5, lane = threadIdx.x & 31;
  int m0 = (blockIdx.x * 8 + wave) * 16;
  v16bf a[4];
#pragma unroll
  for (int ks = 0; ks < 4; ++ks) a[ks] = load_a_f32(h, HD, m0, ks * 32, lane);
#pragma unroll
  for (int nt = 0; nt < 8; ++nt) {
    v8f acc = {};
#pragma unroll
    for (int ks = 0; ks < 4; ++ks)
      acc = wmma_bf(a[ks], load_b_bf16(W, HD, ks * 32, nt * 16, lane), acc);
    int col = nt * 16 + (lane & 15);
    int mb = m0 + (lane >> 4) * 8;
#pragma unroll
    for (int r = 0; r < 8; ++r) xg[(long)(mb + r) * HD + col] = (bf16)acc[r];
  }
}

// ---------------------------------------------------------------------------
// fused edge kernel: one wave per atom (32 consecutive edges)
//   W = ssp(ea @ w1 + b1) @ w2 + b2;  W *= C;  msg = xg[src]*W;
//   agg[atom] = column-sum over the atom's 32 edges (bf16 out)
// All [E,128] intermediates live in registers/LDS only.
// ---------------------------------------------------------------------------
__global__ __launch_bounds__(128) void k_edge_fused(
    const bf16* ea, const float* Cbuf, const int* src, const bf16* xg,
    const bf16* w1, const float* b1, const bf16* w2, const float* b2,
    bf16* agg) {
  __shared__ bf16 sh_hid[4][16 * HD];   // per-wave ssp(GEMM1) tile (A-layout src)
  __shared__ bf16 sh_x[4][KNEI * HD];   // per-wave staged x[src] rows

  prefetch_weights(w1, NGPAD * HD * 2);
  prefetch_weights(w2, HD * HD * 2);

  int wave = threadIdx.x >> 5, lane = threadIdx.x & 31;
  int atom = blockIdx.x * 4 + wave;
  int e0 = atom * KNEI;
  bf16* shx = sh_x[wave];
  bf16* shh = sh_hid[wave];

  // stage the 32 source-atom feature rows (async global->LDS when available)
#if HAVE_ASYNC_LDS
  for (int j = 0; j < KNEI; ++j) {
    int sa = src[e0 + j];
    __builtin_amdgcn_global_load_async_to_lds_b64(
        (gv2i*)(xg + (long)sa * HD + lane * 4),
        (lv2i*)(shx + j * HD + lane * 4), 0, 0);
  }
  __builtin_amdgcn_s_wait_asynccnt(0);
#else
  for (int j = 0; j < KNEI; ++j) {
    int sa = src[e0 + j];
    const uint2* p = (const uint2*)(xg + (long)sa * HD) + lane;
    *((uint2*)(shx + j * HD) + lane) = *p;
  }
#endif
  __syncthreads();

  float colsum[8];
#pragma unroll
  for (int nt = 0; nt < 8; ++nt) colsum[nt] = 0.f;

  for (int mt = 0; mt < 2; ++mt) {
    int row0 = e0 + mt * 16;
    // ---- GEMM1: [16,64] x [64,128] + bias, ssp, -> LDS (bf16) ----
    v16bf aE0 = load_a_bf16(ea, NGPAD, row0, 0, lane);
    v16bf aE1 = load_a_bf16(ea, NGPAD, row0, 32, lane);
#pragma unroll
    for (int nt = 0; nt < 8; ++nt) {
      v8f acc = {};
      acc = wmma_bf(aE0, load_b_bf16(w1, HD, 0, nt * 16, lane), acc);
      acc = wmma_bf(aE1, load_b_bf16(w1, HD, 32, nt * 16, lane), acc);
      int col = nt * 16 + (lane & 15);
      float bv = b1[col];
      int ml = (lane >> 4) * 8;
#pragma unroll
      for (int r = 0; r < 8; ++r)
        shh[(ml + r) * HD + col] = (bf16)sspf(acc[r] + bv);
    }
    __syncthreads();

    // cutoff values for this lane's 8 edge rows
    int mbase = mt * 16 + (lane >> 4) * 8;
    float Cv[8];
#pragma unroll
    for (int r = 0; r < 8; ++r) Cv[r] = Cbuf[e0 + mbase + r];

    // ---- GEMM2: [16,128] x [128,128], modulate, multiply, partial colsum ----
    v16bf aH[4];
#pragma unroll
    for (int ks = 0; ks < 4; ++ks) aH[ks] = load_a_bf16(shh, HD, 0, ks * 32, lane);
#pragma unroll
    for (int nt = 0; nt < 8; ++nt) {
      v8f acc = {};
#pragma unroll
      for (int ks = 0; ks < 4; ++ks)
        acc = wmma_bf(aH[ks], load_b_bf16(w2, HD, ks * 32, nt * 16, lane), acc);
      int col = nt * 16 + (lane & 15);
      float bv = b2[col];
      float cs = 0.f;
#pragma unroll
      for (int r = 0; r < 8; ++r) {
        float Wv = (acc[r] + bv) * Cv[r];
        cs += Wv * (float)shx[(mbase + r) * HD + col];
      }
      colsum[nt] += cs;
    }
    __syncthreads();   // protect shh before next mt overwrites it
  }

  // fold the two half-row groups (lanes 0-15 <-> 16-31), write agg
#pragma unroll
  for (int nt = 0; nt < 8; ++nt)
    colsum[nt] += __shfl_xor(colsum[nt], 16, 32);
  if (lane < 16) {
#pragma unroll
    for (int nt = 0; nt < 8; ++nt)
      agg[(long)atom * HD + nt * 16 + lane] = (bf16)colsum[nt];
  }
}

// ---------------------------------------------------------------------------
// node update: x2 = agg@cf2 + b;  h += ssp(x2)@intW + intB
// (C-layout -> A-layout transpose through LDS between the two GEMMs)
// ---------------------------------------------------------------------------
__global__ __launch_bounds__(256) void k_node_update(
    const bf16* agg, const bf16* cf2, const float* cf2b, const bf16* intw,
    const float* intb, float* h) {
  __shared__ bf16 sh_t[8][16 * HD];
  prefetch_weights(cf2, HD * HD * 2);
  prefetch_weights(intw, HD * HD * 2);
  int wave = threadIdx.x >> 5, lane = threadIdx.x & 31;
  int m0 = (blockIdx.x * 8 + wave) * 16;
  bf16* sht = sh_t[wave];

  v16bf aG[4];
#pragma unroll
  for (int ks = 0; ks < 4; ++ks) aG[ks] = load_a_bf16(agg, HD, m0, ks * 32, lane);
#pragma unroll
  for (int nt = 0; nt < 8; ++nt) {
    v8f acc = {};
#pragma unroll
    for (int ks = 0; ks < 4; ++ks)
      acc = wmma_bf(aG[ks], load_b_bf16(cf2, HD, ks * 32, nt * 16, lane), acc);
    int col = nt * 16 + (lane & 15);
    float bv = cf2b[col];
    int ml = (lane >> 4) * 8;
#pragma unroll
    for (int r = 0; r < 8; ++r)
      sht[(ml + r) * HD + col] = (bf16)sspf(acc[r] + bv);
  }
  __syncthreads();

  v16bf aT[4];
#pragma unroll
  for (int ks = 0; ks < 4; ++ks) aT[ks] = load_a_bf16(sht, HD, 0, ks * 32, lane);
#pragma unroll
  for (int nt = 0; nt < 8; ++nt) {
    v8f acc = {};
#pragma unroll
    for (int ks = 0; ks < 4; ++ks)
      acc = wmma_bf(aT[ks], load_b_bf16(intw, HD, ks * 32, nt * 16, lane), acc);
    int col = nt * 16 + (lane & 15);
    float bv = intb[col];
    int mb = m0 + (lane >> 4) * 8;
#pragma unroll
    for (int r = 0; r < 8; ++r)
      h[(long)(mb + r) * HD + col] += acc[r] + bv;
  }
}

// ---------------------------------------------------------------------------
// readout / reductions (small tail, VALU)
// ---------------------------------------------------------------------------
__global__ void k_readout(const float* h, const float* w1, const float* b1,
                          const float* w2, const float* b2, float* atom_e) {
  int atom = blockIdx.x * blockDim.x + threadIdx.x;
  if (atom >= NATOMS) return;
  const float* hr = h + (long)atom * HD;
  float acc = b2[0];
  for (int j = 0; j < 64; ++j) {
    float s = b1[j];
#pragma unroll 8
    for (int k = 0; k < HD; ++k) s += hr[k] * w1[k * 64 + j];
    acc += sspf(s) * w2[j];
  }
  atom_e[atom] = acc;
}

__global__ void k_graph_sum(const float* atom_e, float* e_graph) {
  __shared__ float sred[256];
  int tid = threadIdx.x;
  sred[tid] = atom_e[blockIdx.x * 256 + tid];
  __syncthreads();
  for (int s = 128; s > 0; s >>= 1) {
    if (tid < s) sred[tid] += sred[tid + s];
    __syncthreads();
  }
  if (tid == 0) e_graph[blockIdx.x] = sred[0];
}

// FiLM bias + final energy:  energy[i] = B * e_graph[i] + sum_j bias_j
__global__ void k_film_energy(const int* dom_ids, const float* dom_emb,
                              const float* fw1, const float* fb1,
                              const float* fw2, const float* fb2,
                              const float* bw, const float* bb,
                              const float* e_graph, float* energy) {
  __shared__ float s_f1[NGRAPH][HD];
  int j = threadIdx.x;                       // one graph per lane, 32 lanes
  int did = dom_ids[j];
  for (int a = 0; a < HD; ++a) {
    float s = fb1[a];
    for (int t = 0; t < 64; ++t) s += dom_emb[did * 64 + t] * fw1[t * HD + a];
    s_f1[j][a] = fmaxf(s, 0.f);
  }
  float bias = 0.f;
  for (int hh = 0; hh < HD; ++hh) bias += bb[hh];
  for (int k = 0; k < HD; ++k) {
    float fk = fb2[k];
    for (int a = 0; a < HD; ++a) fk += s_f1[j][a] * fw2[a * HD + k];
    float uk = 0.f;
    for (int hh = 0; hh < HD; ++hh) uk += bw[k * HD + hh];
    bias += fk * uk;
  }
  float tot = bias;
  for (int off = 16; off; off >>= 1) tot += __shfl_xor(tot, off, 32);
  energy[j] = (float)NGRAPH * e_graph[j] + tot;
}

// ---------------------------------------------------------------------------
// launch
// ---------------------------------------------------------------------------
extern "C" void kernel_launch(void* const* d_in, const int* in_sizes, int n_in,
                              void* d_out, int out_size, void* d_ws,
                              size_t ws_size, hipStream_t stream) {
  (void)in_sizes; (void)n_in; (void)ws_size;
  const float* pos   = (const float*)d_in[0];
  const int*   z     = (const int*)d_in[1];
  const int*   ei    = (const int*)d_in[3];
  const int*   domid = (const int*)d_in[4];
  const float* emb   = (const float*)d_in[5];
  const float* w1f   = (const float*)d_in[6];
  const float* b1    = (const float*)d_in[7];
  const float* w2f   = (const float*)d_in[8];
  const float* b2    = (const float*)d_in[9];
  const float* cf1f  = (const float*)d_in[10];
  const float* cf2f  = (const float*)d_in[11];
  const float* cf2b  = (const float*)d_in[12];
  const float* intf  = (const float*)d_in[13];
  const float* intb  = (const float*)d_in[14];
  const float* ow1   = (const float*)d_in[15];
  const float* ob1   = (const float*)d_in[16];
  const float* ow2   = (const float*)d_in[17];
  const float* ob2   = (const float*)d_in[18];
  const float* demb  = (const float*)d_in[19];
  const float* fw1   = (const float*)d_in[20];
  const float* fb1   = (const float*)d_in[21];
  const float* fw2   = (const float*)d_in[22];
  const float* fb2   = (const float*)d_in[23];
  const float* bw    = (const float*)d_in[26];
  const float* bb    = (const float*)d_in[27];

  // workspace carve-up (all chunks are >=256B aligned by construction)
  char* w = (char*)d_ws;
  bf16*  ea     = (bf16*)w;  w += (size_t)NEDGES * NGPAD * 2;   // 32 MB
  float* Cbuf   = (float*)w; w += (size_t)NEDGES * 4;           // 1 MB
  float* h      = (float*)w; w += (size_t)NATOMS * HD * 4;      // 4 MB
  bf16*  xg     = (bf16*)w;  w += (size_t)NATOMS * HD * 2;      // 2 MB
  bf16*  agg    = (bf16*)w;  w += (size_t)NATOMS * HD * 2;      // 2 MB
  float* atom_e = (float*)w; w += (size_t)NATOMS * 4;
  float* egraph = (float*)w; w += 256;
  bf16*  w1p    = (bf16*)w;  w += (size_t)NLAYERS * NGPAD * HD * 2;
  bf16*  w2p    = (bf16*)w;  w += (size_t)NLAYERS * HD * HD * 2;
  bf16*  cf1p   = (bf16*)w;  w += (size_t)NLAYERS * HD * HD * 2;
  bf16*  cf2p   = (bf16*)w;  w += (size_t)NLAYERS * HD * HD * 2;
  bf16*  intp   = (bf16*)w;  w += (size_t)NLAYERS * HD * HD * 2;

  // 1) zero full output (energy overwritten at the end; rest stays zero)
  k_zero<<<(out_size + 255) / 256, 256, 0, stream>>>((float*)d_out, out_size);

  // 2) weight conversion to bf16 (+ w1 K-padding to 64)
  const int nmat = NLAYERS * HD * HD;
  k_pad_w1<<<(NLAYERS * NGPAD * HD) / 256, 256, 0, stream>>>(w1f, w1p);
  k_convert_bf16<<<(nmat + 255) / 256, 256, 0, stream>>>(w2f, w2p, nmat);
  k_convert_bf16<<<(nmat + 255) / 256, 256, 0, stream>>>(cf1f, cf1p, nmat);
  k_convert_bf16<<<(nmat + 255) / 256, 256, 0, stream>>>(cf2f, cf2p, nmat);
  k_convert_bf16<<<(nmat + 255) / 256, 256, 0, stream>>>(intf, intp, nmat);

  // 3) embedding gather + edge geometry/basis
  k_embed<<<(NATOMS * HD) / 256, 256, 0, stream>>>(z, emb, h);
  k_prep_edges<<<NEDGES / 256, 256, 0, stream>>>(pos, ei, ea, Cbuf);

  // 4) interaction layers
  for (int l = 0; l < NLAYERS; ++l) {
    k_node_gemm<<<NATOMS / 128, 256, 0, stream>>>(h, cf1p + l * HD * HD, xg);
    k_edge_fused<<<NATOMS / 4, 128, 0, stream>>>(
        ea, Cbuf, ei, xg, w1p + l * NGPAD * HD, b1 + l * HD,
        w2p + l * HD * HD, b2 + l * HD, agg);
    k_node_update<<<NATOMS / 128, 256, 0, stream>>>(
        agg, cf2p + l * HD * HD, cf2b + l * HD, intp + l * HD * HD,
        intb + l * HD, h);
  }

  // 5) readout + reductions + FiLM bias -> energy
  k_readout<<<NATOMS / 256, 256, 0, stream>>>(h, ow1, ob1, ow2, ob2, atom_e);
  k_graph_sum<<<NGRAPH, 256, 0, stream>>>(atom_e, egraph);
  k_film_energy<<<1, 32, 0, stream>>>(domid, demb, fw1, fb1, fw2, fb2, bw, bb,
                                      egraph, (float*)d_out);
}
